// PLSTMModel_73469710566021
// MI455X (gfx1250) — matrix operand order repaired and checked by reading, not compiled, not run on hardware
//
#include <hip/hip_runtime.h>

// ---------------------------------------------------------------------------
// Part-aware LSTM (NTU RGB+D) for MI455X / gfx1250.
//   PARTS=5, BATCH=128, SEQ=300, DP=30, HID=512, CLASSES=60
// Strategy:
//  * Recurrence independent per batch row -> 8 persistent workgroups,
//    each owns 16 batch rows for all 300 timesteps. No grid sync.
//  * 512 threads = 16 wave32 per WG (4 waves/SIMD on the WGP) for latency
//    hiding; each wave owns 2 hidden 16-col tiles.
//  * c-state [5][16][512] f32 (160KB) + ping-pong h (bf16) live in LDS
//    (CDNA5 WGP has 320KB LDS).
//  * All GEMMs via v_wmma_f32_16x16x32_bf16 (f32 accumulate).
//  * Weights pre-packed once per launch (f32 -> bf16) into a contiguous-k (k = 16*(lane/16) + e)
//    wave32 B-operand lane layout, same k order as loadA; hot-loop B loads are
//    2x b128 per lane straight from L2 (9MB packed, L2-resident).
//  * x_{t+1} is prefetched with GLOBAL_LOAD_ASYNC_TO_LDS_B32 (ASYNCcnt)
//    into a double-buffered LDS staging area while step t computes.
// ---------------------------------------------------------------------------

#define NPARTS 5
#define NBATCH 128
#define NSEQ   300
#define NDP    30
#define NHID   512
#define NG     1536     // 3*NHID
#define NCLS   60
#define MB     16       // batch rows per workgroup
#define NTHR   512      // 16 wave32

typedef __attribute__((ext_vector_type(16))) __bf16          v16bf;
typedef __attribute__((ext_vector_type(16))) unsigned short  v16u;
typedef __attribute__((ext_vector_type(8)))  unsigned short  v8u;
typedef __attribute__((ext_vector_type(8)))  float           v8f;

// ---- packed-weight layout (unsigned short / bf16 elements) ----------------
// Wh : [P][KC=16][NT=96][lane=32][16]
// Wx : [P][NT=96][lane=32][16]              (K rows 0..29 valid, 30..31 zero)
// Wxo: [KC=5(part)][NT=32][lane=32][16]     (per-part 30 rows + 2 zero pad)
// Who: [KC=16][NT=32][lane=32][16]
#define OFF_WH   ((size_t)0)
#define N_WH     ((size_t)NPARTS*16*96*32*16)      // 3,932,160
#define OFF_WX   (OFF_WH + N_WH)
#define N_WX     ((size_t)NPARTS*96*32*16)         //   245,760
#define OFF_WXO  (OFF_WX + N_WX)
#define N_WXO    ((size_t)5*32*32*16)              //    81,920
#define OFF_WHO  (OFF_WXO + N_WXO)
#define N_WHO    ((size_t)16*32*32*16)             //   262,144
// total = 4,521,984 u16 = 9,043,968 bytes of workspace

static __device__ __forceinline__ unsigned short f2bf(float f) {
  unsigned u = __builtin_bit_cast(unsigned, f);
  unsigned r = u + 0x7FFFu + ((u >> 16) & 1u);     // round-to-nearest-even
  return (unsigned short)(r >> 16);
}
static __device__ __forceinline__ float bf2f(unsigned short h) {
  unsigned u = ((unsigned)h) << 16;
  return __builtin_bit_cast(float, u);
}
static __device__ __forceinline__ float sigf(float x) {
  return 1.0f / (1.0f + __expf(-x));
}
static __device__ __forceinline__ float tnhf(float x) {
  float e = __expf(2.0f * x);
  return 1.0f - 2.0f / (e + 1.0f);                 // safe at +/-inf
}

// A operand (16-bit 16x32): lane<16 -> row=lane, K {0..15};
// lane>=16 -> row=lane-16, K {16..31}. q = &buf[row][kbase].
static __device__ __forceinline__ v16bf loadA(const unsigned short* q, int hiK) {
  v8u lo = *(const v8u*)(q + hiK * 16);
  v8u hi = *(const v8u*)(q + hiK * 16 + 8);
  v16u u = __builtin_shufflevector(lo, hi, 0,1,2,3,4,5,6,7,8,9,10,11,12,13,14,15);
  return __builtin_bit_cast(v16bf, u);
}

// B operand from pre-packed global tile: lane's 16 values are contiguous.
static __device__ __forceinline__ v16bf loadB(const unsigned short* tile, int lane) {
  const unsigned short* q = tile + lane * 16;
  v8u lo = *(const v8u*)(q);
  v8u hi = *(const v8u*)(q + 8);
  v16u u = __builtin_shufflevector(lo, hi, 0,1,2,3,4,5,6,7,8,9,10,11,12,13,14,15);
  return __builtin_bit_cast(v16bf, u);
}

#define WMMA(acc, A, Bv)                                                     \
  acc = __builtin_amdgcn_wmma_f32_16x16x32_bf16(false, (A), false, (Bv),     \
                                                (short)0, (acc), false, false)

// Fire-and-forget prefetch of x_t (16 rows x 5 parts x 30 f32) into the LDS
// staging buffer via the gfx1250 async-to-LDS engine (tracked by ASYNCcnt).
static __device__ __forceinline__ void async_stage_x(
    const float* __restrict__ x, int t, int b0, float* xstg_buf, int tid) {
  for (int i = tid; i < MB * NPARTS * NDP; i += NTHR) {
    int r = i / (NPARTS * NDP);
    int rem = i - r * (NPARTS * NDP);
    int p = rem / NDP, d = rem - p * NDP;
    const float* src =
        x + ((((size_t)p * NBATCH + (b0 + r)) * NSEQ + t) * NDP) + d;
    unsigned dst =
        (unsigned)(size_t)(xstg_buf + (size_t)r * (NPARTS * 32) + p * 32 + d);
    asm volatile("global_load_async_to_lds_b32 %0, %1, off"
                 :: "v"(dst), "v"(src) : "memory");
  }
}
static __device__ __forceinline__ void wait_async0() {
  asm volatile("s_wait_asynccnt 0x0" ::: "memory");
}

// ---------------------------------------------------------------------------
// Pack f32 weights -> bf16 WMMA B-operand layout in workspace.
// One task = one (tile, lane): writes 16 contiguous bf16.
// ---------------------------------------------------------------------------
__global__ __launch_bounds__(256) void plstm_pack_kernel(
    const float* __restrict__ Wx, const float* __restrict__ Wh,
    const float* __restrict__ Wxo, const float* __restrict__ Who,
    unsigned short* __restrict__ ws) {
  const int TWh  = NPARTS * 16 * 96 * 32;   // 245760
  const int TWx  = NPARTS * 96 * 32;        //  15360
  const int TWxo = 5 * 32 * 32;             //   5120
  const int TWho = 16 * 32 * 32;            //  16384
  const int TOT  = TWh + TWx + TWxo + TWho;
  for (int idx = blockIdx.x * 256 + threadIdx.x; idx < TOT;
       idx += gridDim.x * 256) {
    if (idx < TWh) {
      int lane = idx & 31, t = idx >> 5;
      int nt = t % 96, kc = (t / 96) % 16, p = t / (96 * 16);
      int n = nt * 16 + (lane & 15);
      unsigned short* dp = ws + OFF_WH + (size_t)idx * 16;
      for (int e = 0; e < 16; ++e) {
        int k = kc * 32 + (lane >> 4) * 16 + e;
        dp[e] = f2bf(Wh[((size_t)p * NHID + k) * NG + n]);
      }
    } else if (idx < TWh + TWx) {
      int i = idx - TWh;
      int lane = i & 31, t = i >> 5;
      int nt = t % 96, p = t / 96;
      int n = nt * 16 + (lane & 15);
      unsigned short* dp = ws + OFF_WX + (size_t)i * 16;
      for (int e = 0; e < 16; ++e) {
        int k = (lane >> 4) * 16 + e;                     // padded K 0..31
        float v = (k < NDP) ? Wx[((size_t)p * NDP + k) * NG + n] : 0.0f;
        dp[e] = f2bf(v);
      }
    } else if (idx < TWh + TWx + TWxo) {
      int i = idx - TWh - TWx;
      int lane = i & 31, t = i >> 5;
      int nt = t % 32, kc = t / 32;                        // kc == part
      int n = nt * 16 + (lane & 15);
      unsigned short* dp = ws + OFF_WXO + (size_t)i * 16;
      for (int e = 0; e < 16; ++e) {
        int kk = (lane >> 4) * 16 + e;                     // within-part 0..31
        float v = (kk < NDP) ? Wxo[((size_t)kc * NDP + kk) * NHID + n] : 0.0f;
        dp[e] = f2bf(v);
      }
    } else {
      int i = idx - TWh - TWx - TWxo;
      int lane = i & 31, t = i >> 5;
      int nt = t % 32, kc = t / 32;
      int n = nt * 16 + (lane & 15);
      unsigned short* dp = ws + OFF_WHO + (size_t)i * 16;
      for (int e = 0; e < 16; ++e) {
        int k = kc * 32 + (lane >> 4) * 16 + e;
        dp[e] = f2bf(Who[(size_t)k * NHID + n]);
      }
    }
  }
}

// ---------------------------------------------------------------------------
// Persistent per-batch-tile LSTM. Block = 512 threads (16 wave32), grid = 8.
// ---------------------------------------------------------------------------
__global__ __launch_bounds__(NTHR, 1) void plstm_main_kernel(
    const float* __restrict__ x,      // [P][B][T][D]
    const float* __restrict__ bias,   // [P][3H]
    const float* __restrict__ bo,     // [H]
    const float* __restrict__ fcw,    // [H][CLS]
    const float* __restrict__ fcb,    // [CLS]
    const unsigned short* __restrict__ wpk,
    float* __restrict__ out) {        // [B][CLS]
  __shared__ float          cst[NPARTS][MB][NHID];        // 160 KB
  __shared__ unsigned short hbuf[2][MB][NHID];            //  32 KB (bf16)
  __shared__ float          xstg[2][MB][NPARTS * 32];     //  20 KB (f32, async)
  __shared__ unsigned short xbuf[MB][NPARTS * 32];        //   5 KB (bf16)
  __shared__ float          lbuf[MB][64];                 //   4 KB

  const int tid  = threadIdx.x;
  const int lane = tid & 31;
  const int wv   = tid >> 5;                // 0..15
  const int nloc = lane & 15;
  const int hiK  = lane >> 4;
  const int b0   = blockIdx.x * MB;

  const unsigned short* pWh  = wpk + OFF_WH;
  const unsigned short* pWx  = wpk + OFF_WX;
  const unsigned short* pWxo = wpk + OFF_WXO;
  const unsigned short* pWho = wpk + OFF_WHO;

  // zero-init state (also zeroes the pad columns d=30,31 of xstg once;
  // async copies only ever write d<30, so pads stay zero)
  for (int i = tid; i < NPARTS * MB * NHID; i += NTHR) (&cst[0][0][0])[i] = 0.0f;
  for (int i = tid; i < 2 * MB * NHID; i += NTHR) (&hbuf[0][0][0])[i] = 0;
  for (int i = tid; i < 2 * MB * NPARTS * 32; i += NTHR) (&xstg[0][0][0])[i] = 0.0f;
  __syncthreads();

  // prefetch x_0 into staging buffer 0
  async_stage_x(x, 0, b0, &xstg[0][0][0], tid);

  int cur = 0;
#pragma unroll 1
  for (int t = 0; t < NSEQ; ++t) {
    // retire this step's async x staging (own wave), then sync all waves
    wait_async0();
    __syncthreads();
    // convert staged f32 -> bf16 A-operand buffer
    for (int i = tid; i < MB * NPARTS * 32; i += NTHR)
      (&xbuf[0][0])[i] = f2bf((&xstg[t & 1][0][0])[i]);
    // kick off next step's x prefetch into the other staging buffer
    if (t + 1 < NSEQ)
      async_stage_x(x, t + 1, b0, &xstg[(t + 1) & 1][0][0], tid);
    __syncthreads();

    const unsigned short* hcur = &hbuf[cur][0][0];

#pragma unroll 1
    for (int tt = 0; tt < 2; ++tt) {
      const int j   = wv + tt * 16;             // hidden 16-col tile, 0..31
      const int col = j * 16 + nloc;
      v8f csum = {0, 0, 0, 0, 0, 0, 0, 0};

#pragma unroll 1
      for (int p = 0; p < NPARTS; ++p) {
        v8f ai = {0,0,0,0,0,0,0,0}, af = {0,0,0,0,0,0,0,0}, ag = {0,0,0,0,0,0,0,0};
        {  // x-gate K chunk (one 16x16x32)
          v16bf A  = loadA(&xbuf[nloc][p * 32], hiK);
          v16bf Bi = loadB(pWx + ((size_t)p * 96 + j)        * 512, lane);
          v16bf Bf = loadB(pWx + ((size_t)p * 96 + 32 + j)   * 512, lane);
          v16bf Bg = loadB(pWx + ((size_t)p * 96 + 64 + j)   * 512, lane);
          WMMA(ai, A, Bi); WMMA(af, A, Bf); WMMA(ag, A, Bg);
        }
#pragma unroll 4
        for (int kc = 0; kc < 16; ++kc) {       // h-gate: 512 = 16 x 32
          v16bf A = loadA(hcur + nloc * NHID + kc * 32, hiK);
          size_t base = ((size_t)p * 16 + kc) * 96;
          v16bf Bi = loadB(pWh + (base + j)      * 512, lane);
          v16bf Bf = loadB(pWh + (base + 32 + j) * 512, lane);
          v16bf Bg = loadB(pWh + (base + 64 + j) * 512, lane);
          WMMA(ai, A, Bi); WMMA(af, A, Bf); WMMA(ag, A, Bg);
        }
        // per-part cell update; this wave exclusively owns these columns
#pragma unroll
        for (int r = 0; r < 8; ++r) {
          int m = r + hiK * 8;
          float iv = sigf(ai[r] + bias[(size_t)p * NG + col]);
          float fv = sigf(af[r] + bias[(size_t)p * NG + 512 + col]);
          float gv = tnhf(ag[r] + bias[(size_t)p * NG + 1024 + col]);
          float cn = fv * cst[p][m][col] + iv * gv;
          cst[p][m][col] = cn;
          csum[r] += cn;
        }
      }

      // shared output gate on same hidden tile (uses OLD h)
      v8f ao = {0, 0, 0, 0, 0, 0, 0, 0};
#pragma unroll
      for (int kc = 0; kc < 5; ++kc) {
        v16bf A  = loadA(&xbuf[nloc][kc * 32], hiK);
        v16bf Bo = loadB(pWxo + ((size_t)kc * 32 + j) * 512, lane);
        WMMA(ao, A, Bo);
      }
#pragma unroll 4
      for (int kc = 0; kc < 16; ++kc) {
        v16bf A  = loadA(hcur + nloc * NHID + kc * 32, hiK);
        v16bf Bo = loadB(pWho + ((size_t)kc * 32 + j) * 512, lane);
        WMMA(ao, A, Bo);
      }
#pragma unroll
      for (int r = 0; r < 8; ++r) {
        int m = r + hiK * 8;
        float ov = sigf(ao[r] + bo[col]);
        float hv = ov * tnhf(csum[r]);
        hbuf[cur ^ 1][m][col] = f2bf(hv);
      }
    }
    __syncthreads();
    cur ^= 1;
  }

  // epilogue: logits + log-softmax (tiny: 16x60 per WG)
  for (int i = tid; i < MB * NCLS; i += NTHR) {
    int r = i / NCLS, c = i % NCLS;
    float acc = fcb[c];
    for (int k = 0; k < NHID; ++k)
      acc += bf2f(hbuf[cur][r][k]) * fcw[(size_t)k * NCLS + c];
    lbuf[r][c] = acc;
  }
  __syncthreads();
  if (tid < MB) {
    int r = tid;
    float mx = -3.0e38f;
    for (int c = 0; c < NCLS; ++c) mx = fmaxf(mx, lbuf[r][c]);
    float s = 0.0f;
    for (int c = 0; c < NCLS; ++c) s += __expf(lbuf[r][c] - mx);
    float lse = mx + __logf(s);
    for (int c = 0; c < NCLS; ++c)
      out[(size_t)(b0 + r) * NCLS + c] = lbuf[r][c] - lse;
  }
}

// ---------------------------------------------------------------------------
extern "C" void kernel_launch(void* const* d_in, const int* in_sizes, int n_in,
                              void* d_out, int out_size, void* d_ws,
                              size_t ws_size, hipStream_t stream) {
  const float* x_in = (const float*)d_in[0];   // [P][B][T][D]
  const float* Wx   = (const float*)d_in[1];   // [P][D][3H]
  const float* Wh   = (const float*)d_in[2];   // [P][H][3H]
  const float* b    = (const float*)d_in[3];   // [P][3H]
  const float* Wxo  = (const float*)d_in[4];   // [IN][H]
  const float* Who  = (const float*)d_in[5];   // [H][H]
  const float* bo   = (const float*)d_in[6];   // [H]
  const float* fcw  = (const float*)d_in[7];   // [H][CLS]
  const float* fcb  = (const float*)d_in[8];   // [CLS]
  unsigned short* ws = (unsigned short*)d_ws;  // needs ~9.1 MB

  // 1) pack weights f32 -> bf16 WMMA B layout (L2-resident afterwards)
  plstm_pack_kernel<<<1104, 256, 0, stream>>>(Wx, Wh, Wxo, Who, ws);
  // 2) persistent batch-parallel LSTM: 8 WGs x 16 batch rows x 300 steps
  plstm_main_kernel<<<NBATCH / MB, NTHR, 0, stream>>>(
      x_in, b, bo, fcw, fcb, ws, (float*)d_out);
}
